// Pool_33277406609549
// MI455X (gfx1250) — compile-verified, matrix operand order checked by reading, not run on hardware
//
#include <hip/hip_runtime.h>

// ---------------------------------------------------------------------------
// Problem constants (match reference)
// ---------------------------------------------------------------------------
constexpr int kDIMS = 3;
constexpr int kB    = 256;
constexpr int kN    = 256;
constexpr int kK    = 16;
constexpr int kHID  = 64;
constexpr int kF    = 64;
constexpr int kEPG  = 8192;
constexpr int kNTOT = kB * kN;          // 65536
constexpr int kETOT = kB * kEPG;        // 2097152
constexpr float kEPS   = 1e-15f;
constexpr float kScale = (float)kK / (float)kN;   // 0.0625

// d_out flat layout (reference tuple order: px, pei, pea, pbatch, loss, s)
constexpr int OFF_PX   = 0;
constexpr int OFF_PEI  = OFF_PX  + kB * kK * kF * kDIMS;   // 786432
constexpr int OFF_PEA  = OFF_PEI + 2 * kB * kK * kK;       // 917504
constexpr int OFF_PB   = OFF_PEA + kB * kK * kK * kDIMS;   // 1114112
constexpr int OFF_LOSS = OFF_PB  + kB * kK;                // 1118208
constexpr int OFF_S    = OFF_LOSS + 1;                     // 1118209

// ---------------------------------------------------------------------------
// LDS layout (dynamic shared). Persistent region + 128KB reusable scratch.
// ---------------------------------------------------------------------------
constexpr int L_ADJ  = 0;                 // 128*256*4 = 131072 (scratch region)
constexpr int L_HX   = 131072;            // 256*64 bf16 = 32768 (input x, row-major)
constexpr int L_H0   = L_HX  + 32768;
constexpr int L_H1   = L_H0  + 32768;
constexpr int L_H2   = L_H1  + 32768;     // 256*16 bf16 = 8192
constexpr int L_AGG  = L_H2  + 8192;      // 256*64 bf16 = 32768
constexpr int L_HT   = L_AGG + 32768;     // 64*256 bf16 = 32768 (current h, transposed)
constexpr int SMEM_BYTES = L_HT + 32768;  // 303104  (< 320KB WGP LDS)
// aliases carved inside the 128KB scratch region (time-disjoint, sync'd):
constexpr int L_WMT  = 0;                 // 64*64 bf16 (Wm transposed)
constexpr int L_WST  = 8192;
constexpr int L_BIAS = 16384;             // 64 f32
constexpr int L_FC1  = 0;                 // 144*50 f32 = 28800
constexpr int L_FC2  = 28800;             // 50*16 f32 = 3200
constexpr int L_FB1  = 32000;             // 50 f32
constexpr int L_FB2  = 32256;             // 16 f32
constexpr int L_SMX  = 40960;             // 256*16 f32 = 16384 (softmax s)
constexpr int L_TKM  = 57344;             // 256*16 f32 (tmp = s^T A, [m][k])
constexpr int L_RED  = 73728;             // 2 f32 (ent, mod partials)

// ---------------------------------------------------------------------------
// Types / helpers
// ---------------------------------------------------------------------------
typedef float    v8f   __attribute__((ext_vector_type(8)));
typedef float    v4f   __attribute__((ext_vector_type(4)));
typedef unsigned v8u   __attribute__((ext_vector_type(8)));
typedef __bf16   bf16;
typedef bf16     v16bf __attribute__((ext_vector_type(16)));
typedef bf16     v8bf  __attribute__((ext_vector_type(8)));

__device__ __forceinline__ bf16 f2bf(float f) {
  unsigned u = __builtin_bit_cast(unsigned, f);
  unsigned r = u + 0x7FFFu + ((u >> 16) & 1u);   // round-to-nearest-even
  unsigned short h = (unsigned short)(r >> 16);
  return __builtin_bit_cast(bf16, h);
}

__device__ __forceinline__ v8f wmma_bf16(v16bf a, v16bf b, v8f c) {
  // D(16x16,f32) = A(16x32,bf16) * B(32x16,bf16) + C
  return __builtin_amdgcn_wmma_f32_16x16x32_bf16(false, a, false, b,
                                                 (short)0, c, false, false);
}

// CDNA5 16-bit A-matrix (16x32) lane layout: lane L holds row M=L%16;
// element e: K = (lane>=16 ? 8 : 0) + (e<8 ? e : 8+e)  => K 0..7 and 16..23
// are CONTIGUOUS per lane: fragment = two 16-byte LDS loads.
__device__ __forceinline__ v16bf load_frag_bf16(const bf16* base, int ld,
                                                int m0, int k0, int lane) {
  const int hi = lane >> 4;
  const bf16* p = base + (m0 + (lane & 15)) * ld + k0 + hi * 8;
  v8bf lo = *(const v8bf*)p;          // K = k0+hi*8 .. +7
  v8bf hh = *(const v8bf*)(p + 16);   // K = k0+hi*8+16 .. +23
  return __builtin_shufflevector(lo, hh, 0, 1, 2, 3, 4, 5, 6, 7,
                                 8, 9, 10, 11, 12, 13, 14, 15);
}

// A-fragment from an f32 LDS matrix (adjacency): 16 scalar u32 loads (two
// contiguous 32B runs; backend re-merges them into ds_load_b128) and integer
// high-half packing (truncating f32->bf16; adjacency is a sum of uniform(0,1)
// edge weights so RNE is unnecessary). Scalar loads + explicit integer uses
// keep every element live in the IR (a v4f-load + perm-intrinsic formulation
// was observed to mis-fold elements into duplicates).
__device__ __forceinline__ v16bf load_frag_f32(const float* base, int ld,
                                               int m0, int k0, int lane) {
  const int hi = lane >> 4;
  const unsigned* p =
      (const unsigned*)(base + (m0 + (lane & 15)) * ld + k0 + hi * 8);
  v8u pk;
#pragma unroll
  for (int j = 0; j < 4; ++j) {
    unsigned e0 = p[2 * j];                 // elems 0..7: K = k0+hi*8 + 0..7
    unsigned e1 = p[2 * j + 1];
    unsigned g0 = p[16 + 2 * j];            // elems 8..15: K = +16 .. +23
    unsigned g1 = p[16 + 2 * j + 1];
    pk[j]     = (e0 >> 16) | (e1 & 0xFFFF0000u);   // {hi(e1), hi(e0)}
    pk[4 + j] = (g0 >> 16) | (g1 & 0xFFFF0000u);
  }
  return __builtin_bit_cast(v16bf, pk);
}

// C/D (16x16,f32): lane L holds column N=L%16; VGPR r holds M = r + 8*(L>=16)
__device__ __forceinline__ void store_d_bf16(bf16* base, int ld,
                                             int m0, int n0, v8f acc, int lane) {
  const int hi = lane >> 4;
  const int n = n0 + (lane & 15);
#pragma unroll
  for (int r = 0; r < 8; ++r)
    base[(m0 + r + hi * 8) * ld + n] = f2bf(acc[r]);
}

// ---------------------------------------------------------------------------
// Mega-kernel: one workgroup per (graph b, dim d). 256 threads = 8 wave32.
// ---------------------------------------------------------------------------
__global__ __launch_bounds__(256) void mega_kernel(
    const float* __restrict__ x,    const float* __restrict__ eattr,
    const float* __restrict__ xtp,
    const float* __restrict__ Wm0,  const float* __restrict__ Ws0,
    const float* __restrict__ b0v,
    const float* __restrict__ Wm1,  const float* __restrict__ Ws1,
    const float* __restrict__ b1v,
    const float* __restrict__ Wm2,  const float* __restrict__ Ws2,
    const float* __restrict__ b2v,
    const float* __restrict__ fcW1, const float* __restrict__ fcb1,
    const float* __restrict__ fcW2, const float* __restrict__ fcb2,
    const int*   __restrict__ ei,
    float* __restrict__ out, float* __restrict__ acc_ws)
{
  extern __shared__ char smem[];
  const int tid  = threadIdx.x;
  const int wave = tid >> 5;
  const int lane = tid & 31;
  const int b = blockIdx.x;
  const int d = blockIdx.y;

  float* adjF = (float*)(smem + L_ADJ);
  bf16*  hX   = (bf16*)(smem + L_HX);
  bf16*  h0   = (bf16*)(smem + L_H0);
  bf16*  h1   = (bf16*)(smem + L_H1);
  bf16*  h2   = (bf16*)(smem + L_H2);
  bf16*  agg  = (bf16*)(smem + L_AGG);
  bf16*  hT   = (bf16*)(smem + L_HT);

  // ---- stage x[b] as bf16, row-major + transposed ----
  {
    const float* xb = x + (size_t)b * kN * kHID;
    for (int i = tid; i < kN * kHID; i += 256) {
      int n = i & 255, c = i >> 8;                 // c in [0,64)
      bf16 v = f2bf(xb[n * kHID + c]);
      hX[n * kHID + c] = v;
      hT[c * kN + n]   = v;
    }
  }

  const float* Wm0d = Wm0 + d * kHID * kHID;
  const float* Ws0d = Ws0 + d * kHID * kHID;
  const float* Wm1d = Wm1 + d * kHID * kHID;
  const float* Ws1d = Ws1 + d * kHID * kHID;
  const float* Wm2d = Wm2 + d * kHID * kK;
  const float* Ws2d = Ws2 + d * kHID * kK;
  const float* b0d  = b0v + d * kHID;
  const float* b1d  = b1v + d * kHID;
  const float* b2d  = b2v + d * kK;

  const int eb = b * kEPG;   // edges of graph b are contiguous

  // =========================== GCN layers ================================
#pragma unroll
  for (int L = 0; L < 3; ++L) {
    bf16* hin  = (L == 0) ? hX : (L == 1) ? h0 : h1;     // constant-folded
    bf16* hout = (L == 0) ? h0 : (L == 1) ? h1 : h2;
    const float* Wm = (L == 0) ? Wm0d : (L == 1) ? Wm1d : Wm2d;
    const float* Ws = (L == 0) ? Ws0d : (L == 1) ? Ws1d : Ws2d;
    const float* bb = (L == 0) ? b0d  : (L == 1) ? b1d  : b2d;
    const int Wout = (L == 2) ? kK : kHID;

    // ---- SpMM agg = A * hin, two 128-row halves of the LDS adjacency ----
#pragma unroll 1
    for (int rh = 0; rh < 2; ++rh) {
      __syncthreads();                           // region reuse guard
      {
        v4f z = {0.f, 0.f, 0.f, 0.f};
        v4f* az = (v4f*)adjF;
        for (int i = tid; i < 128 * 256 / 4; i += 256) az[i] = z;
      }
      __syncthreads();
      for (int e = eb + tid; e < eb + kEPG; e += 256) {
        int rl = ei[e] & 255;                    // row % N (N=256)
        if ((rl >> 7) == rh) {
          int cl = ei[kETOT + e] & 255;
          atomicAdd(&adjF[(rl & 127) * 256 + cl], eattr[e * 3 + d]);
        }
      }
      __syncthreads();

      // wave w owns rows [w*16, w*16+16) of this half; K = 256, Nout = 64
      v8f acc[4] = {};
#pragma unroll
      for (int kt = 0; kt < 8; ++kt) {
        v16bf af = load_frag_f32(adjF, 256, wave * 16, kt * 32, lane);
#pragma unroll
        for (int nt = 0; nt < 4; ++nt) {
          // B[k][n] = hin[k][n] = hT[n][k]  (two 16B LDS loads)
          v16bf bf = load_frag_bf16(hT, kN, nt * 16, kt * 32, lane);
          acc[nt] = wmma_bf16(af, bf, acc[nt]);
        }
      }
      const int rowbase = rh * 128 + wave * 16;
#pragma unroll
      for (int nt = 0; nt < 4; ++nt)
        store_d_bf16(agg, kHID, rowbase, nt * 16, acc[nt], lane);
    }
    __syncthreads();

    // ---- dense: hout = relu(agg*Wm + hin*Ws + bias) ----
    bf16*  WmT   = (bf16*)(smem + L_WMT);        // [n][k], transposed
    bf16*  WsT   = (bf16*)(smem + L_WST);
    float* biasS = (float*)(smem + L_BIAS);
    for (int i = tid; i < kHID * Wout; i += 256) {
      int kk = i / Wout, nn = i % Wout;          // Wout constant -> folded
      WmT[nn * kHID + kk] = f2bf(Wm[i]);
      WsT[nn * kHID + kk] = f2bf(Ws[i]);
    }
    if (tid < Wout) biasS[tid] = bb[tid];
    __syncthreads();

    const int nT = Wout >> 4;                    // 4,4,1
    for (int t = wave; t < 16 * nT; t += 8) {
      int mt = t / nT, nt = t % nT;
      v8f acc = {};
#pragma unroll
      for (int kt = 0; kt < 2; ++kt) {
        acc = wmma_bf16(load_frag_bf16(agg, kHID, mt * 16, kt * 32, lane),
                        load_frag_bf16(WmT, kHID, nt * 16, kt * 32, lane), acc);
        acc = wmma_bf16(load_frag_bf16(hin, kHID, mt * 16, kt * 32, lane),
                        load_frag_bf16(WsT, kHID, nt * 16, kt * 32, lane), acc);
      }
      const int hi = lane >> 4;
      const int n  = nt * 16 + (lane & 15);
      const float bv = biasS[n];
      v8bf tcol;
#pragma unroll
      for (int r = 0; r < 8; ++r) {
        float v = acc[r] + bv;
        v = v > 0.f ? v : 0.f;
        bf16 bvv = f2bf(v);
        hout[(mt * 16 + r + hi * 8) * Wout + n] = bvv;
        tcol[r] = bvv;
      }
      if (L < 2) {
        // transposed copy for next layer's SpMM B operand: one 16B store
        *(v8bf*)(hT + n * kN + mt * 16 + hi * 8) = tcol;
      }
    }
    __syncthreads();
  }

  // =================== FC + softmax + entropy (VALU) =====================
  float* fc1s = (float*)(smem + L_FC1);
  float* fc2s = (float*)(smem + L_FC2);
  float* fb1s = (float*)(smem + L_FB1);
  float* fb2s = (float*)(smem + L_FB2);
  float* Sld  = (float*)(smem + L_SMX);
  float* tkm  = (float*)(smem + L_TKM);
  float* red  = (float*)(smem + L_RED);

  for (int i = tid; i < 144 * 50; i += 256) fc1s[i] = fcW1[i];
  for (int i = tid; i < 50 * 16;  i += 256) fc2s[i] = fcW2[i];
  if (tid < 50) fb1s[tid] = fcb1[tid];
  if (tid < 16) fb2s[tid] = fcb2[tid];
  if (tid < 2)  red[tid]  = 0.f;
  __syncthreads();

  {
    const int n = tid;                            // one node per thread
    float z[50];
    for (int j = 0; j < 50; ++j) {
      float a = fb1s[j];
      // rotate start index by n to avoid 64-bank LDS conflicts
      for (int cc = 0; cc < 64; ++cc) {
        int c = (cc + n) & 63;
        a += (float)h0[n * 64 + c] * fc1s[c * 50 + j];
      }
      for (int cc = 0; cc < 64; ++cc) {
        int c = (cc + n) & 63;
        a += (float)h1[n * 64 + c] * fc1s[(64 + c) * 50 + j];
      }
      for (int cc = 0; cc < 16; ++cc) {
        int c = (cc + n) & 15;
        a += (float)h2[n * 16 + c] * fc1s[(128 + c) * 50 + j];
      }
      z[j] = a;
    }
    float lg[16];
    float mx = -3.4e38f;
    for (int k2 = 0; k2 < 16; ++k2) {
      float a = fb2s[k2];
      for (int j = 0; j < 50; ++j) a += z[j] * fc2s[j * 16 + k2];
      lg[k2] = a;
      mx = a > mx ? a : mx;
    }
    float ssum = 0.f;
    for (int k2 = 0; k2 < 16; ++k2) {
      float e2 = __expf(lg[k2] - mx);
      lg[k2] = e2;
      ssum += e2;
    }
    const float inv = 1.f / ssum;
    float ent = 0.f;
    for (int k2 = 0; k2 < 16; ++k2) {
      float sv = lg[k2] * inv;
      Sld[n * 16 + k2] = sv;
      out[OFF_S + (((size_t)d * kB + b) * kN + n) * kK + k2] = sv;
      ent += -sv * __logf(sv + kEPS);
    }
    atomicAdd(&red[0], ent);
  }
  for (int i = tid; i < kN * kK; i += 256) tkm[i] = 0.f;
  __syncthreads();

  // ========================= px = scale * s^T x ==========================
  for (int idx = tid; idx < kK * kF; idx += 256) {
    int k2 = idx >> 6, f = idx & 63;
    const float* xp = xtp + (size_t)(b * kN) * kF * kDIMS + f * kDIMS + d;
    float s0 = 0.f;
    for (int n2 = 0; n2 < kN; ++n2)
      s0 += Sld[n2 * 16 + k2] * xp[(size_t)n2 * kF * kDIMS];
    out[OFF_PX + ((b * kK + k2) * kF + f) * kDIMS + d] = s0 * kScale;
  }

  // ======== edge pass: tmp[m][k] += w*s[rl][k] (s^T A), + modularity ======
  float modloc = 0.f;
  for (int e = eb + tid; e < eb + kEPG; e += 256) {
    int rl = ei[e] & 255;
    int cl = ei[kETOT + e] & 255;
    float w = eattr[e * 3 + d];
    float dd = 0.f;
#pragma unroll
    for (int k2 = 0; k2 < 16; ++k2) {
      float sr = Sld[rl * 16 + k2];
      float df = sr - Sld[cl * 16 + k2];
      dd += df * df;
      atomicAdd(&tkm[cl * 16 + k2], w * sr);
    }
    modloc += w * dd;
  }
  atomicAdd(&red[1], modloc);
  __syncthreads();

  // ================== padj = scale^2 * (s^T A) s  → pea ==================
  {
    int k2 = tid >> 4, l = tid & 15;
    float a = 0.f;
    for (int m = 0; m < kN; ++m)
      a += tkm[m * 16 + k2] * Sld[m * 16 + l];
    out[OFF_PEA + ((b * kK + k2) * kK + l) * kDIMS + d] = a * (kScale * kScale);
  }
  if (tid == 0) {
    atomicAdd(&acc_ws[0], red[1]);   // modularity sum
    atomicAdd(&acc_ws[1], red[0]);   // entropy sum
  }
}

// ---------------------------------------------------------------------------
// init: zero the loss accumulators (harness does not re-poison ws per replay)
// ---------------------------------------------------------------------------
__global__ void init_kernel(float* ws) {
  if (threadIdx.x < 2) ws[threadIdx.x] = 0.f;
}

// ---------------------------------------------------------------------------
// finalize: deterministic pei/pbatch patterns + scalar loss
// ---------------------------------------------------------------------------
__global__ void fin_kernel(const float* __restrict__ ws, float* __restrict__ out) {
  const int idx = blockIdx.x * 256 + threadIdx.x;
  const int TOTPEI = 2 * kB * kK * kK;           // 131072
  if (idx < TOTPEI) {
    int half = idx / (kB * kK * kK);
    int rr   = idx % (kB * kK * kK);
    int b    = rr / (kK * kK);
    int j    = rr % (kK * kK);
    int v    = b * kK + (half == 0 ? (j / kK) : (j % kK));
    out[OFF_PEI + idx] = (float)v;
  } else if (idx < TOTPEI + kB * kK) {
    int i = idx - TOTPEI;
    out[OFF_PB + i] = (float)(i >> 4);
  } else if (idx == TOTPEI + kB * kK) {
    out[OFF_LOSS] = ws[0] / (float)kETOT + ws[1] / (float)(kDIMS * kNTOT);
  }
}

// ---------------------------------------------------------------------------
extern "C" void kernel_launch(void* const* d_in, const int* in_sizes, int n_in,
                              void* d_out, int out_size, void* d_ws, size_t ws_size,
                              hipStream_t stream) {
  (void)in_sizes; (void)n_in; (void)out_size; (void)ws_size;
  const float* x    = (const float*)d_in[0];
  const float* ea   = (const float*)d_in[1];
  const float* xtp  = (const float*)d_in[2];
  const float* Wm0  = (const float*)d_in[3];
  const float* Ws0  = (const float*)d_in[4];
  const float* b0v  = (const float*)d_in[5];
  const float* Wm1  = (const float*)d_in[6];
  const float* Ws1  = (const float*)d_in[7];
  const float* b1v  = (const float*)d_in[8];
  const float* Wm2  = (const float*)d_in[9];
  const float* Ws2  = (const float*)d_in[10];
  const float* b2v  = (const float*)d_in[11];
  const float* fcW1 = (const float*)d_in[12];
  const float* fcb1 = (const float*)d_in[13];
  const float* fcW2 = (const float*)d_in[14];
  const float* fcb2 = (const float*)d_in[15];
  const int*   ei   = (const int*)d_in[16];
  float* out = (float*)d_out;
  float* ws  = (float*)d_ws;

  (void)hipFuncSetAttribute(reinterpret_cast<const void*>(mega_kernel),
                            hipFuncAttributeMaxDynamicSharedMemorySize,
                            SMEM_BYTES);

  init_kernel<<<1, 64, 0, stream>>>(ws);

  dim3 grid(kB, kDIMS);
  mega_kernel<<<grid, 256, SMEM_BYTES, stream>>>(
      x, ea, xtp, Wm0, Ws0, b0v, Wm1, Ws1, b1v, Wm2, Ws2, b2v,
      fcW1, fcb1, fcW2, fcb2, ei, out, ws);

  const int finTot = 2 * kB * kK * kK + kB * kK + 1;
  fin_kernel<<<(finTot + 255) / 256, 256, 0, stream>>>(ws, out);
}
